// EGSCStudent_71768903516484
// MI455X (gfx1250) — compile-verified, matrix operand order checked by reading, not run on hardware
//
#include <hip/hip_runtime.h>
#include <hip/hip_bf16.h>
#include <math.h>

typedef __attribute__((ext_vector_type(16))) _Float16 v16h;
typedef __attribute__((ext_vector_type(8)))  _Float16 v8h;
typedef __attribute__((ext_vector_type(8)))  float    v8f;

union HV { v16h v; v8h h[2]; };

#define NG_CONST 1000
#define BN_EPS 1e-5f

// ---------------------------------------------------------------------------
// generic zero-fill
// ---------------------------------------------------------------------------
__global__ void zero_kernel(float* __restrict__ p, int n) {
    int i = blockIdx.x * blockDim.x + threadIdx.x;
    if (i < n) p[i] = 0.0f;
}

// ---------------------------------------------------------------------------
// edge aggregation: agg[dst] += x[src]   (C floats per node, C % 4 == 0)
// one thread per (edge, 4-float chunk): float4 gather + 4 atomic adds
// ---------------------------------------------------------------------------
template<int C>
__global__ void edge_agg_kernel(const float* __restrict__ x,
                                const int* __restrict__ src,
                                const int* __restrict__ dst,
                                float* __restrict__ agg, int e) {
    constexpr int NCH = C / 4;
    int t  = blockIdx.x * blockDim.x + threadIdx.x;
    int ei = t / NCH;
    if (ei >= e) return;
    int c  = (t % NCH) * 4;
    int s = src[ei], d = dst[ei];
    const float4 v = *(const float4*)(x + (size_t)s * C + c);
    float* ap = agg + (size_t)d * C + c;
    atomicAdd(ap + 0, v.x);
    atomicAdd(ap + 1, v.y);
    atomicAdd(ap + 2, v.z);
    atomicAdd(ap + 3, v.w);
}

// ---------------------------------------------------------------------------
// fused GIN MLP:  h = relu(((1+eps)*x + agg) @ W1 + b1) @ W2 + b2
// writes raw h (pre-BN) and accumulates per-feature sum / sumsq for BN.
// One wave32 owns a 16-row tile; both GEMMs via v_wmma_f32_16x16x32_f16.
// Weights are pre-swizzled into fragment-major LDS so each lane's WMMA
// operand is two contiguous 16B LDS vectors (ds_load_b128).
// sums layout: sums[0..F-1] = column sums, sums[F..2F-1] = sum-of-squares
// ---------------------------------------------------------------------------
template<int CIN, int F>
__global__ __launch_bounds__(256)
void gin_mlp_kernel(const float* __restrict__ x, const float* __restrict__ agg,
                    const float* __restrict__ epsp,
                    const float* __restrict__ W1, const float* __restrict__ b1,
                    const float* __restrict__ W2, const float* __restrict__ b2,
                    float* __restrict__ hraw, float* __restrict__ sums,
                    int nrows) {
    constexpr int K2  = (F < 32) ? 32 : F;          // padded K for second GEMM
    constexpr int NT  = F / 16;                     // N tiles
    constexpr int NB1 = (CIN / 32) * NT;            // B fragments, GEMM1
    constexpr int NB2 = (K2 / 32) * NT;             // B fragments, GEMM2
    __shared__ _Float16 w1f[NB1 * 512];             // fragment-major: [frag][lane][16]
    __shared__ _Float16 w2f[NB2 * 512];
    __shared__ _Float16 h1s[8][16 * F];             // per-wave staging tile

    const int tid = threadIdx.x;
    // cooperative load + f32->f16 convert + swizzle into fragment-major layout
    for (int i = tid; i < NB1 * 512; i += 256) {
        int e    = i & 15;
        int ln   = (i >> 4) & 31;
        int fidx = i >> 9;
        int nt   = fidx % NT, kb = fidx / NT;
        int base = (ln >= 16) ? 8 : 0;
        int k = kb * 32 + ((e >> 3) << 4) + base + (e & 7);
        int n = nt * 16 + (ln & 15);
        w1f[i] = (_Float16)W1[k * F + n];
    }
    for (int i = tid; i < NB2 * 512; i += 256) {
        int e    = i & 15;
        int ln   = (i >> 4) & 31;
        int fidx = i >> 9;
        int nt   = fidx % NT, kb = fidx / NT;
        int base = (ln >= 16) ? 8 : 0;
        int k = kb * 32 + ((e >> 3) << 4) + base + (e & 7);
        int n = nt * 16 + (ln & 15);
        w2f[i] = (k < F) ? (_Float16)W2[k * F + n] : (_Float16)0.0f;
    }
    __syncthreads();

    const int wave = tid >> 5, lane = tid & 31;
    const int tile = blockIdx.x * 8 + wave;
    if (tile * 16 >= nrows) return;

    const float epsv  = 1.0f + epsp[0];
    const int   m     = lane & 15;             // A-matrix row owned by lane
    const int   row   = tile * 16 + m;
    const int   kbase = (lane >= 16) ? 8 : 0;  // A/B 16-bit K striping
    const int   n     = lane & 15;             // C-matrix column owned by lane
    const int   half  = lane >> 4;             // C-matrix M-half

    const float* xr = x   + (size_t)row * CIN;
    const float* ar = agg + (size_t)row * CIN;
    // stream-ahead prefetch (global_prefetch_b8), ~8 tiles ahead
    __builtin_prefetch(xr + 128 * CIN, 0, 1);
    __builtin_prefetch(ar + 128 * CIN, 0, 1);

    // ---------------- GEMM1: acc = Ztile @ W1 ----------------
    v8f acc[NT];
#pragma unroll
    for (int nt = 0; nt < NT; nt++) acc[nt] = v8f{0,0,0,0,0,0,0,0};

#pragma unroll
    for (int kk = 0; kk < CIN; kk += 32) {
        HV a;
#pragma unroll
        for (int g = 0; g < 2; g++) {
            const int k0 = kk + g * 16 + kbase;      // multiple of 8 floats
            const float4 x0 = *(const float4*)(xr + k0);
            const float4 x1 = *(const float4*)(xr + k0 + 4);
            const float4 a0 = *(const float4*)(ar + k0);
            const float4 a1 = *(const float4*)(ar + k0 + 4);
            v8h t;
            t[0] = (_Float16)(epsv * x0.x + a0.x);
            t[1] = (_Float16)(epsv * x0.y + a0.y);
            t[2] = (_Float16)(epsv * x0.z + a0.z);
            t[3] = (_Float16)(epsv * x0.w + a0.w);
            t[4] = (_Float16)(epsv * x1.x + a1.x);
            t[5] = (_Float16)(epsv * x1.y + a1.y);
            t[6] = (_Float16)(epsv * x1.z + a1.z);
            t[7] = (_Float16)(epsv * x1.w + a1.w);
            a.h[g] = t;
        }
#pragma unroll
        for (int nt = 0; nt < NT; nt++) {
            const v8h* bp = (const v8h*)(w1f + (((kk >> 5) * NT + nt) * 32 + lane) * 16);
            HV b; b.h[0] = bp[0]; b.h[1] = bp[1];
            acc[nt] = __builtin_amdgcn_wmma_f32_16x16x32_f16(
                false, a.v, false, b.v, (short)0, acc[nt], false, false);
        }
    }

    // bias + relu, stage h1 (f16) to per-wave LDS for C->A layout transpose
    _Float16* hs = h1s[wave];
#pragma unroll
    for (int nt = 0; nt < NT; nt++) {
#pragma unroll
        for (int r = 0; r < 8; r++) {
            float v = acc[nt][r] + b1[nt * 16 + n];
            v = v > 0.0f ? v : 0.0f;
            hs[(half * 8 + r) * F + nt * 16 + n] = (_Float16)v;
        }
    }
    // wave-internal LDS ordering (CDNA5 split counters): stores -> loads
    asm volatile("s_wait_dscnt 0" ::: "memory");

    // ---------------- GEMM2: acc2 = relu(h1) @ W2 ----------------
    v8f acc2[NT];
#pragma unroll
    for (int nt = 0; nt < NT; nt++) acc2[nt] = v8f{0,0,0,0,0,0,0,0};

#pragma unroll
    for (int kk = 0; kk < K2; kk += 32) {
        HV a;
#pragma unroll
        for (int g = 0; g < 2; g++) {
            const int k0 = kk + g * 16 + kbase;      // multiple of 8 halves
            if (k0 < F) {
                a.h[g] = *(const v8h*)(hs + m * F + k0);   // ds_load_b128
            } else {
                v8h z;
#pragma unroll
                for (int q = 0; q < 8; q++) z[q] = (_Float16)0.0f;
                a.h[g] = z;
            }
        }
#pragma unroll
        for (int nt = 0; nt < NT; nt++) {
            const v8h* bp = (const v8h*)(w2f + (((kk >> 5) * NT + nt) * 32 + lane) * 16);
            HV b; b.h[0] = bp[0]; b.h[1] = bp[1];
            acc2[nt] = __builtin_amdgcn_wmma_f32_16x16x32_f16(
                false, a.v, false, b.v, (short)0, acc2[nt], false, false);
        }
    }

    // epilogue: bias add, write raw h, accumulate BN column sums/sumsq
#pragma unroll
    for (int nt = 0; nt < NT; nt++) {
        float s = 0.0f, q = 0.0f;
#pragma unroll
        for (int r = 0; r < 8; r++) {
            float v = acc2[nt][r] + b2[nt * 16 + n];
            hraw[(size_t)(tile * 16 + half * 8 + r) * F + nt * 16 + n] = v;
            s += v;
            q += v * v;
        }
        s += __shfl_xor(s, 16, 32);   // combine the two 8-row halves of column n
        q += __shfl_xor(q, 16, 32);
        if (lane < 16) {
            atomicAdd(&sums[nt * 16 + n],     s);
            atomicAdd(&sums[F + nt * 16 + n], q);
        }
    }
}

// ---------------------------------------------------------------------------
// BatchNorm finalize (batch statistics):  out = g*(h-mu)*rsqrt(var+eps)+b
// ---------------------------------------------------------------------------
template<int F>
__global__ void bn_finalize_kernel(const float* __restrict__ hraw,
                                   const float* __restrict__ sums,
                                   const float* __restrict__ gamma,
                                   const float* __restrict__ beta,
                                   float* __restrict__ out, int nrows) {
    int idx = blockIdx.x * blockDim.x + threadIdx.x;
    if (idx >= nrows * F) return;
    int f = idx % F;
    float invn = 1.0f / (float)nrows;
    float mu   = sums[f] * invn;
    float var  = sums[F + f] * invn - mu * mu;
    out[idx] = gamma[f] * (hraw[idx] - mu) * rsqrtf(var + BN_EPS) + beta[f];
}

// ---------------------------------------------------------------------------
// readout stage 1: per-graph node count + feature sum (F3 = 16)
// ---------------------------------------------------------------------------
__global__ void seg_sum_kernel(const float* __restrict__ x,
                               const int* __restrict__ batch,
                               float* __restrict__ cnt,
                               float* __restrict__ summ, int n) {
    int i = blockIdx.x * blockDim.x + threadIdx.x;
    if (i >= n) return;
    int b = batch[i];
    atomicAdd(&cnt[b], 1.0f);
    const float* xr = x + (size_t)i * 16;
#pragma unroll
    for (int f = 0; f < 16; f++) atomicAdd(&summ[b * 16 + f], xr[f]);
}

// readout stage 2: c = sigmoid(mean @ W + b)    [NG x 16]
__global__ void att_c_kernel(const float* __restrict__ cnt,
                             const float* __restrict__ summ,
                             const float* __restrict__ W,
                             const float* __restrict__ b,
                             float* __restrict__ c) {
    int t = blockIdx.x * blockDim.x + threadIdx.x;
    if (t >= NG_CONST * 16) return;
    int g = t / 16, f = t % 16;
    float inv = 1.0f / fmaxf(cnt[g], 1.0f);
    float acc = b[f];
#pragma unroll
    for (int k = 0; k < 16; k++) acc += (summ[g * 16 + k] * inv) * W[k * 16 + f];
    c[t] = 1.0f / (1.0f + expf(-acc));
}

// readout stage 3: gated add-pool  hg[b] += sigmoid(<x_i, c_b>) * x_i
__global__ void gate_pool_kernel(const float* __restrict__ x,
                                 const int* __restrict__ batch,
                                 const float* __restrict__ c,
                                 float* __restrict__ hg, int n) {
    int i = blockIdx.x * blockDim.x + threadIdx.x;
    if (i >= n) return;
    int b = batch[i];
    const float* xr = x + (size_t)i * 16;
    const float* cr = c + b * 16;
    float dot = 0.0f;
#pragma unroll
    for (int f = 0; f < 16; f++) dot += xr[f] * cr[f];
    float gate = 1.0f / (1.0f + expf(-dot));
#pragma unroll
    for (int f = 0; f < 16; f++) atomicAdd(&hg[b * 16 + f], gate * xr[f]);
}

// ---------------------------------------------------------------------------
// EFN + final FC head (NG = 1000 graphs; one thread per graph)
// ---------------------------------------------------------------------------
__device__ __forceinline__
void efn_one(const float* a, const float* b,
             const float* aW1, const float* ab1,
             const float* aW2, const float* ab2,
             const float* mW,  const float* mb, float* outv) {
    float h[32];
#pragma unroll
    for (int k = 0; k < 16; k++) { h[k] = a[k]; h[16 + k] = b[k]; }
    float r[8];
#pragma unroll
    for (int o = 0; o < 8; o++) {
        float ac = ab1[o];
#pragma unroll
        for (int k = 0; k < 32; k++) ac += h[k] * aW1[k * 8 + o];
        r[o] = fmaxf(ac, 0.0f);
    }
#pragma unroll
    for (int k = 0; k < 32; k++) {
        float ac = ab2[k];
#pragma unroll
        for (int o = 0; o < 8; o++) ac += r[o] * aW2[o * 32 + k];
        float at = tanhf(ac);
        h[k] = at * h[k] + h[k];
    }
#pragma unroll
    for (int f = 0; f < 16; f++) {
        float ac = mb[f];
#pragma unroll
        for (int k = 0; k < 32; k++) ac += h[k] * mW[k * 16 + f];
        outv[f] = fmaxf(ac, 0.0f);
    }
}

__global__ void efn_final_kernel(const float* __restrict__ hi,
                                 const float* __restrict__ hj,
                                 const float* aW1, const float* ab1,
                                 const float* aW2, const float* ab2,
                                 const float* mW,  const float* mb,
                                 const float* fW1, const float* fb1,
                                 const float* fW2, const float* fb2,
                                 float* __restrict__ out) {
    int g = blockIdx.x * blockDim.x + threadIdx.x;
    if (g >= NG_CONST) return;
    float hAB[16], hAA[16], hBB[16];
    efn_one(hi + g * 16, hj + g * 16, aW1, ab1, aW2, ab2, mW, mb, hAB);
    efn_one(hi + g * 16, hi + g * 16, aW1, ab1, aW2, ab2, mW, mb, hAA);
    efn_one(hj + g * 16, hj + g * 16, aW1, ab1, aW2, ab2, mW, mb, hBB);
    float t[8];
#pragma unroll
    for (int o = 0; o < 8; o++) {
        float ac = fb1[o];
#pragma unroll
        for (int k = 0; k < 16; k++) ac += hAB[k] * fW1[k * 8 + o];
        t[o] = fmaxf(ac, 0.0f);
    }
    float sc = fb2[0];
#pragma unroll
    for (int k = 0; k < 8; k++) sc += t[k] * fW2[k];
    out[g] = sc;
#pragma unroll
    for (int f = 0; f < 16; f++) {
        out[NG_CONST + g * 16 + f]                 = hAB[f] - hBB[f];  // h_Ab
        out[NG_CONST + NG_CONST * 16 + g * 16 + f] = hAB[f] - hAA[f];  // h_aB
    }
}

// ---------------------------------------------------------------------------
// host orchestration
// ---------------------------------------------------------------------------
static inline int cdiv(int a, int b) { return (a + b - 1) / b; }

extern "C" void kernel_launch(void* const* d_in, const int* in_sizes, int n_in,
                              void* d_out, int out_size, void* d_ws, size_t ws_size,
                              hipStream_t stream) {
    const int N = in_sizes[0] / 64;   // 200000
    const int E = in_sizes[2] / 2;    // 3200000

    const float* x_in[2]  = { (const float*)d_in[0], (const float*)d_in[1] };
    const int*   ei[2]    = { (const int*)d_in[2],   (const int*)d_in[3]   };
    const int*   batch[2] = { (const int*)d_in[4],   (const int*)d_in[5]   };

    const float* eps1  = (const float*)d_in[6];
    const float* g1W1  = (const float*)d_in[7];
    const float* g1b1  = (const float*)d_in[8];
    const float* g1W2  = (const float*)d_in[9];
    const float* g1b2  = (const float*)d_in[10];
    const float* g1g   = (const float*)d_in[11];
    const float* g1bt  = (const float*)d_in[12];
    const float* eps2  = (const float*)d_in[13];
    const float* g2W1  = (const float*)d_in[14];
    const float* g2b1  = (const float*)d_in[15];
    const float* g2W2  = (const float*)d_in[16];
    const float* g2b2  = (const float*)d_in[17];
    const float* g2g   = (const float*)d_in[18];
    const float* g2bt  = (const float*)d_in[19];
    const float* eps3  = (const float*)d_in[20];
    const float* g3W1  = (const float*)d_in[21];
    const float* g3b1  = (const float*)d_in[22];
    const float* g3W2  = (const float*)d_in[23];
    const float* g3b2  = (const float*)d_in[24];
    const float* g3g   = (const float*)d_in[25];
    const float* g3bt  = (const float*)d_in[26];
    const float* attW  = (const float*)d_in[27];
    const float* attb  = (const float*)d_in[28];
    const float* aW1   = (const float*)d_in[29];
    const float* ab1   = (const float*)d_in[30];
    const float* aW2   = (const float*)d_in[31];
    const float* ab2   = (const float*)d_in[32];
    const float* mW    = (const float*)d_in[33];
    const float* mb    = (const float*)d_in[34];
    const float* fW1   = (const float*)d_in[35];
    const float* fb1   = (const float*)d_in[36];
    const float* fW2   = (const float*)d_in[37];
    const float* fb2   = (const float*)d_in[38];

    // workspace partition (floats)
    float* ws    = (float*)d_ws;
    size_t N64   = (size_t)N * 64;
    float* xA    = ws;                 // N*64
    float* xB    = xA + N64;           // N*64
    float* agg   = xB + N64;           // N*64
    float* hraw  = agg + N64;          // N*64
    float* sums  = hraw + N64;         // 128
    float* cnt   = sums + 128;         // NG
    float* summ  = cnt + NG_CONST;     // NG*16
    float* cgate = summ + NG_CONST*16; // NG*16
    float* hg[2];
    hg[0] = cgate + NG_CONST * 16;     // NG*16
    hg[1] = hg[0] + NG_CONST * 16;     // NG*16

    const int TB = 256;
    const int mlp_blocks = cdiv(cdiv(N, 16), 8);

    for (int s = 0; s < 2; s++) {
        const int* src = ei[s];
        const int* dst = ei[s] + E;

        // ---- GIN layer 1: CIN=64 -> F=64 ----
        zero_kernel<<<cdiv(N * 64, TB), TB, 0, stream>>>(agg, N * 64);
        edge_agg_kernel<64><<<cdiv(E * 16, TB), TB, 0, stream>>>(x_in[s], src, dst, agg, E);
        zero_kernel<<<1, 128, 0, stream>>>(sums, 128);
        gin_mlp_kernel<64, 64><<<mlp_blocks, TB, 0, stream>>>(
            x_in[s], agg, eps1, g1W1, g1b1, g1W2, g1b2, hraw, sums, N);
        bn_finalize_kernel<64><<<cdiv(N * 64, TB), TB, 0, stream>>>(hraw, sums, g1g, g1bt, xA, N);

        // ---- GIN layer 2: CIN=64 -> F=32 ----
        zero_kernel<<<cdiv(N * 64, TB), TB, 0, stream>>>(agg, N * 64);
        edge_agg_kernel<64><<<cdiv(E * 16, TB), TB, 0, stream>>>(xA, src, dst, agg, E);
        zero_kernel<<<1, 128, 0, stream>>>(sums, 128);
        gin_mlp_kernel<64, 32><<<mlp_blocks, TB, 0, stream>>>(
            xA, agg, eps2, g2W1, g2b1, g2W2, g2b2, hraw, sums, N);
        bn_finalize_kernel<32><<<cdiv(N * 32, TB), TB, 0, stream>>>(hraw, sums, g2g, g2bt, xB, N);

        // ---- GIN layer 3: CIN=32 -> F=16 ----
        zero_kernel<<<cdiv(N * 32, TB), TB, 0, stream>>>(agg, N * 32);
        edge_agg_kernel<32><<<cdiv(E * 8, TB), TB, 0, stream>>>(xB, src, dst, agg, E);
        zero_kernel<<<1, 128, 0, stream>>>(sums, 128);
        gin_mlp_kernel<32, 16><<<mlp_blocks, TB, 0, stream>>>(
            xB, agg, eps3, g3W1, g3b1, g3W2, g3b2, hraw, sums, N);
        bn_finalize_kernel<16><<<cdiv(N * 16, TB), TB, 0, stream>>>(hraw, sums, g3g, g3bt, xA, N);

        // ---- global attention readout ----
        zero_kernel<<<cdiv(NG_CONST, TB), TB, 0, stream>>>(cnt, NG_CONST);
        zero_kernel<<<cdiv(NG_CONST * 16, TB), TB, 0, stream>>>(summ, NG_CONST * 16);
        seg_sum_kernel<<<cdiv(N, TB), TB, 0, stream>>>(xA, batch[s], cnt, summ, N);
        att_c_kernel<<<cdiv(NG_CONST * 16, TB), TB, 0, stream>>>(cnt, summ, attW, attb, cgate);
        zero_kernel<<<cdiv(NG_CONST * 16, TB), TB, 0, stream>>>(hg[s], NG_CONST * 16);
        gate_pool_kernel<<<cdiv(N, TB), TB, 0, stream>>>(xA, batch[s], cgate, hg[s], N);
    }

    // ---- EFN + final FC head ----
    efn_final_kernel<<<cdiv(NG_CONST, TB), TB, 0, stream>>>(
        hg[0], hg[1], aW1, ab1, aW2, ab2, mW, mb, fW1, fb1, fW2, fb2, (float*)d_out);
}
